// multi_head_trnasformer_self_attn_16432544874607
// MI455X (gfx1250) — matrix operand order repaired and checked by reading, not compiled, not run on hardware
//
#include <hip/hip_runtime.h>
#include <stdint.h>

// ---------------- problem dims (compile-time) ----------------
#define EDIM 512
#define TSEQ 2048
#define NH   8
#define NL   6
#define FFD  2048
#define VOC  32000
#define BS   2
#define HDIM 64
#define MTOK (BS*TSEQ)   // 4096

typedef unsigned short u16;
typedef unsigned int   u32;
typedef unsigned long long u64;

typedef __attribute__((ext_vector_type(16))) __bf16 v16bf;
typedef __attribute__((ext_vector_type(8)))  float  v8f;
typedef __attribute__((ext_vector_type(4)))  unsigned int u32x4;
typedef __attribute__((ext_vector_type(8)))  int          i32x8;
typedef __attribute__((ext_vector_type(4)))  int          i32x4;

union Frag16 { uint4 q[2]; v16bf v; };

__device__ __forceinline__ u16 f2bf(float f) {
    u32 u = __float_as_uint(f);
    u32 r = (u + 0x7FFFu + ((u >> 16) & 1u)) >> 16;   // RNE f32 -> bf16
    return (u16)r;
}

__device__ __forceinline__ float rmax16(float v) {   // reduce within 16-lane half
    v = fmaxf(v, __shfl_xor(v, 1, 32));
    v = fmaxf(v, __shfl_xor(v, 2, 32));
    v = fmaxf(v, __shfl_xor(v, 4, 32));
    v = fmaxf(v, __shfl_xor(v, 8, 32));
    return v;
}
__device__ __forceinline__ float rsum16(float v) {
    v += __shfl_xor(v, 1, 32);
    v += __shfl_xor(v, 2, 32);
    v += __shfl_xor(v, 4, 32);
    v += __shfl_xor(v, 8, 32);
    return v;
}

// ---- Tensor Data Mover: 2D tile (tileW x tileH elements of 2 bytes) -> LDS ----
// Descriptor per CDNA5 ISA.
// padIntervalCode/padAmountCode: LDS padding applied by the DMA itself.
__device__ __forceinline__ void tdm_load_2d_bf16(
    const u16* gptr, u32 lds_byte_off, int tileW, int tileH,
    int rowStrideElems, int tensorW, int tensorH,
    int padIntervalCode, int padAmountCode)
{
    const u64 ga = (u64)(uintptr_t)gptr;
    u32x4 g0;
    g0.x = 1u;                                  // count=1, user mode, no gather
    g0.y = lds_byte_off;                        // lds_addr
    g0.z = (u32)ga;                             // global_addr[31:0]
    g0.w = (u32)(ga >> 32) | (2u << 30);        // global_addr[56:32] | type=2
    i32x8 g1;
    g1[0] = (1 << 16)                           // data_size = 1 -> 2 bytes
          | (1 << 20)                           // pad_enable
          | (padIntervalCode << 22)
          | (padAmountCode << 25);
    g1[1] = (tensorW & 0xFFFF) << 16;                           // tensor_dim0[15:0]
    g1[2] = ((u32)tensorW >> 16) | ((tensorH & 0xFFFF) << 16);  // dim0 hi | dim1 lo
    g1[3] = ((u32)tensorH >> 16) | ((u32)tileW << 16);          // dim1 hi | tile_dim0
    g1[4] = tileH;                              // tile_dim1 (tile_dim2 = 0)
    g1[5] = rowStrideElems;                     // tensor_dim0_stride[31:0]
    g1[6] = 0;                                  // stride hi | dim1_stride lo
    g1[7] = 0;
    i32x4 g2 = {};
    i32x4 g3 = {};
#if defined(__clang_major__) && (__clang_major__ >= 23)
    i32x8 g4 = {};
    __builtin_amdgcn_tensor_load_to_lds(g0, g1, g2, g3, g4, 0);
#else
    __builtin_amdgcn_tensor_load_to_lds(g0, g1, g2, g3, 0);
#endif
}

// ---------------- embedding: h = tok_emb[x] + pos_emb ----------------
__global__ __launch_bounds__(256) void embed_kernel(
    const int* __restrict__ x, const float* __restrict__ tok,
    const float* __restrict__ pos, float* __restrict__ h)
{
    int idx = blockIdx.x * 256 + threadIdx.x;           // over MTOK*EDIM
    int e  = idx & (EDIM - 1);
    int bt = idx >> 9;                                  // EDIM = 512
    int t  = bt & (TSEQ - 1);
    h[idx] = tok[(size_t)x[bt] * EDIM + e] + pos[t * EDIM + e];
}

// ---------------- weight transpose + f32->bf16: Wt[N][K] = W[K][N] ----------------
__global__ __launch_bounds__(256) void wconvt_kernel(
    const float* __restrict__ W, u16* __restrict__ Wt, int K, int N)
{
    size_t idx = (size_t)blockIdx.x * 256 + threadIdx.x;  // over K*N
    int n = (int)(idx % N);
    int k = (int)(idx / N);
    Wt[(size_t)n * K + k] = f2bf(W[idx]);
}

// ---------------- f32 -> bf16 elementwise ----------------
__global__ __launch_bounds__(256) void conv_kernel(
    const float* __restrict__ a, u16* __restrict__ b)
{
    int i = blockIdx.x * 256 + threadIdx.x;
    b[i] = f2bf(a[i]);
}

// ---------------- LayerNorm: one wave per row of E=512, bf16 out ----------------
__global__ __launch_bounds__(256) void layernorm_kernel(
    const float* __restrict__ h, const float* __restrict__ sc,
    const float* __restrict__ bi, u16* __restrict__ y)
{
    int row  = blockIdx.x * 8 + (threadIdx.x >> 5);
    int lane = threadIdx.x & 31;
    const float* r = h + (size_t)row * EDIM;

    float4 xv[4];
    float s = 0.f, sq = 0.f;
#pragma unroll
    for (int j = 0; j < 4; ++j) {
        xv[j] = *(const float4*)(r + j * 128 + lane * 4);
        s  += xv[j].x + xv[j].y + xv[j].z + xv[j].w;
        sq += xv[j].x*xv[j].x + xv[j].y*xv[j].y + xv[j].z*xv[j].z + xv[j].w*xv[j].w;
    }
#pragma unroll
    for (int m = 16; m >= 1; m >>= 1) { s += __shfl_xor(s, m, 32); sq += __shfl_xor(sq, m, 32); }
    float mean = s * (1.f / EDIM);
    float var  = sq * (1.f / EDIM) - mean * mean;
    float inv  = rsqrtf(var + 1e-5f);
#pragma unroll
    for (int j = 0; j < 4; ++j) {
        int e0 = j * 128 + lane * 4;
        float4 sv = *(const float4*)(sc + e0);
        float4 bv = *(const float4*)(bi + e0);
        u16* o = y + (size_t)row * EDIM + e0;
        o[0] = f2bf((xv[j].x - mean) * inv * sv.x + bv.x);
        o[1] = f2bf((xv[j].y - mean) * inv * sv.y + bv.y);
        o[2] = f2bf((xv[j].z - mean) * inv * sv.z + bv.z);
        o[3] = f2bf((xv[j].w - mean) * inv * sv.w + bv.w);
    }
}

// ---------------- WMMA GEMM with TDM-staged, double-buffered A tiles ----------------
// C[M,N] = act(A[M,K]bf16 @ Wt[N,K]bf16^T + bias) (+resid)
// OUT_MODE: 0 = f32 out, 1 = bf16 out, 2 = bf16 out, V-transposed per head [B,H,HD,T]
// A tile: 128 rows x 64 cols bf16 (128B/row); TDM pads rows to 144B in LDS.
#define KSTEP   64
#define A_ROWB  144                       // 128B data + 16B pad per row
#define A_BUFSZ (128 * A_ROWB)            // 18432 B per buffer

template<int OUT_MODE, bool RELU, bool RESID>
__global__ __launch_bounds__(256) void gemm_bf16_kernel(
    const u16* __restrict__ A, const u16* __restrict__ Bt,
    const float* __restrict__ bias, const float* resid,
    float* outF, u16* outB, int M, int N, int K)
{
    const int nbn = N >> 7;
    const int m0 = (blockIdx.x / nbn) << 7;
    const int n0 = (blockIdx.x % nbn) << 7;
    const int tid  = threadIdx.x;
    const int lane = tid & 31;
    const int wv   = tid >> 5;
    const int wm   = wv >> 2;        // 0..1 -> 64 rows
    const int wn   = wv & 3;         // 0..3 -> 32 cols
    const int hl   = lane >> 4;      // half-wave
    const int l16  = lane & 15;

    // Only static shared object in this kernel -> LDS offset 0.
    __shared__ __align__(16) unsigned char ldsA[2][A_BUFSZ];

    v8f acc[4][2] = {};

    const int nk = K >> 6;           // K is a multiple of 64 here
    // pad codes: interval 4 -> every 128B, amount 3 -> 4 DWORDs (16B)
    if (wv == 0)
        tdm_load_2d_bf16(A + (size_t)m0 * K, 0u, KSTEP, 128, K, K, M, 4, 3);

    for (int s = 0; s < nk; ++s) {
        if (wv == 0) {
            if (s + 1 < nk) {
                tdm_load_2d_bf16(A + (size_t)m0 * K + (s + 1) * KSTEP,
                                 (u32)(((s + 1) & 1) * A_BUFSZ),
                                 KSTEP, 128, K, K, M, 4, 3);
                __builtin_amdgcn_s_wait_tensorcnt((short)1);   // stage s complete
            } else {
                __builtin_amdgcn_s_wait_tensorcnt((short)0);
            }
        }
        __syncthreads();

        const unsigned char* base = ldsA[s & 1];
        const int kbase = s * KSTEP;

        // prefetch next-stage weight fragments into L2/L0
        if (s + 1 < nk) {
            const u16* pf0 = Bt + (size_t)(n0 + wn * 32 + l16) * K + kbase + KSTEP + hl * 16;
            __builtin_prefetch(pf0, 0, 0);
            __builtin_prefetch(pf0 + 16 * K, 0, 0);
        }

#pragma unroll
        for (int kk = 0; kk < 2; ++kk) {                 // two K=32 sub-steps
            Frag16 bf[2];
#pragma unroll
            for (int ni = 0; ni < 2; ++ni) {
                const u16* p = Bt + (size_t)(n0 + wn * 32 + ni * 16 + l16) * K
                             + kbase + kk * 32 + hl * 16;
                bf[ni].q[0] = *(const uint4*)p;
                bf[ni].q[1] = *(const uint4*)(p + 8);
            }
#pragma unroll
            for (int mi = 0; mi < 4; ++mi) {
                Frag16 af;
                const unsigned char* p = base + (wm * 64 + mi * 16 + l16) * A_ROWB
                                       + kk * 64 + hl * 32;
                af.q[0] = *(const uint4*)p;
                af.q[1] = *(const uint4*)(p + 16);
#pragma unroll
                for (int ni = 0; ni < 2; ++ni)
                    acc[mi][ni] = __builtin_amdgcn_wmma_f32_16x16x32_bf16(
                        false, af.v, false, bf[ni].v, (short)0, acc[mi][ni], false, false);
            }
        }
        __syncthreads();   // all waves done reading buf[s&1] before it is overwritten
    }

    // epilogue: C layout -> VGPR i holds row (i + 8*hl), lane%16 holds column
#pragma unroll
    for (int mi = 0; mi < 4; ++mi) {
#pragma unroll
        for (int ni = 0; ni < 2; ++ni) {
            const int col = n0 + wn * 32 + ni * 16 + l16;
            const float bc = bias[col];
#pragma unroll
            for (int i = 0; i < 8; ++i) {
                const int row = m0 + wm * 64 + mi * 16 + i + hl * 8;
                float v = acc[mi][ni][i] + bc;
                if (RELU)  v = fmaxf(v, 0.f);
                if (RESID) v += resid[(size_t)row * N + col];
                if (OUT_MODE == 0) {
                    outF[(size_t)row * N + col] = v;
                } else if (OUT_MODE == 1) {
                    outB[(size_t)row * N + col] = f2bf(v);
                } else { // V transposed per head: [B,H,HD,T]
                    int bb = row >> 11, tp = row & (TSEQ - 1);
                    int hh = col >> 6,  d  = col & (HDIM - 1);
                    outB[(size_t)((bb * NH + hh) * HDIM + d) * TSEQ + tp] = f2bf(v);
                }
            }
        }
    }
}

// ---------------- flash attention: one wave per 16-row q tile ----------------
// q,k: [B,T,H,HD] bf16 ; vt: [B,H,HD,T] bf16 ; o: [B,T,H,HD] bf16
__global__ __launch_bounds__(128) void attn_kernel(
    const u16* __restrict__ q, const u16* __restrict__ k,
    const u16* __restrict__ vt, u16* __restrict__ o)
{
    __shared__ __align__(16) u16 pbuf[4][16 * 32];
    const int wv   = threadIdx.x >> 5;
    const int lane = threadIdx.x & 31;
    const int hl   = lane >> 4;
    const int l16  = lane & 15;

    const int id = blockIdx.x * 4 + wv;
    const int qt = id & 127;             // T/16
    const int hh = (id >> 7) & (NH - 1);
    const int bb = id >> 10;
    const int q0 = qt * 16;

    // Q fragments (A layout), K-dim = HD = 64 -> two K=32 fragments
    Frag16 qf[2];
    {
        const u16* p = q + ((size_t)(bb * TSEQ + q0 + l16) * NH + hh) * HDIM + hl * 16;
        qf[0].q[0] = *(const uint4*)p;        qf[0].q[1] = *(const uint4*)(p + 8);
        qf[1].q[0] = *(const uint4*)(p + 32); qf[1].q[1] = *(const uint4*)(p + 40);
    }

    v8f o_acc[4] = {};
    float mrow[8], lrow[8];
#pragma unroll
    for (int i = 0; i < 8; ++i) { mrow[i] = -1e30f; lrow[i] = 0.f; }

    const int kend = q0 + 16;
    for (int kk0 = 0; kk0 < kend; kk0 += 32) {
        // S = Q @ K^T for two 16-col tiles
        v8f s[2] = {};
#pragma unroll
        for (int j = 0; j < 2; ++j) {
            Frag16 kf0, kf1;
            const u16* kp = k + ((size_t)(bb * TSEQ + kk0 + j * 16 + l16) * NH + hh) * HDIM + hl * 16;
            kf0.q[0] = *(const uint4*)kp;        kf0.q[1] = *(const uint4*)(kp + 8);
            kf1.q[0] = *(const uint4*)(kp + 32); kf1.q[1] = *(const uint4*)(kp + 40);
            s[j] = __builtin_amdgcn_wmma_f32_16x16x32_bf16(false, qf[0].v, false, kf0.v, (short)0, s[j], false, false);
            s[j] = __builtin_amdgcn_wmma_f32_16x16x32_bf16(false, qf[1].v, false, kf1.v, (short)0, s[j], false, false);
        }

        // online softmax (rows live per-VGPR-index within each half-wave)
#pragma unroll
        for (int i = 0; i < 8; ++i) {
            const int row = q0 + i + hl * 8;
            const int c0 = kk0 + l16, c1 = kk0 + 16 + l16;
            float a0 = (c0 <= row) ? s[0][i] * 0.125f : -1e30f;
            float a1 = (c1 <= row) ? s[1][i] * 0.125f : -1e30f;
            float mx = rmax16(fmaxf(a0, a1));
            float mn = fmaxf(mrow[i], mx);
            float al = __expf(mrow[i] - mn);
            float p0 = __expf(a0 - mn);
            float p1 = __expf(a1 - mn);
            s[0][i] = p0; s[1][i] = p1;
            float ps = rsum16(p0 + p1);
            lrow[i] = lrow[i] * al + ps;
            mrow[i] = mn;
#pragma unroll
            for (int t = 0; t < 4; ++t) o_acc[t][i] *= al;
        }

        // transpose P (C layout) -> A layout via per-wave LDS buffer
#pragma unroll
        for (int j = 0; j < 2; ++j)
#pragma unroll
            for (int i = 0; i < 8; ++i)
                pbuf[wv][(i + hl * 8) * 32 + j * 16 + l16] = f2bf(s[j][i]);
        __asm volatile("" ::: "memory");   // wave-synchronous; DS in-order per wave
        Frag16 pf;
        {
            const u16* pp = &pbuf[wv][l16 * 32 + hl * 16];
            pf.q[0] = *(const uint4*)pp;
            pf.q[1] = *(const uint4*)(pp + 8);
        }

        // O += P @ V  (V fragments contiguous thanks to [B,H,HD,T] layout)
#pragma unroll
        for (int t = 0; t < 4; ++t) {
            Frag16 vf;
            const u16* vp = vt + ((size_t)((bb * NH + hh) * HDIM) + t * 16 + l16) * TSEQ + kk0 + hl * 16;
            vf.q[0] = *(const uint4*)vp;
            vf.q[1] = *(const uint4*)(vp + 8);
            o_acc[t] = __builtin_amdgcn_wmma_f32_16x16x32_bf16(false, pf.v, false, vf.v, (short)0, o_acc[t], false, false);
        }
    }

    // normalize + store
#pragma unroll
    for (int i = 0; i < 8; ++i) {
        const float inv = 1.f / lrow[i];
        const int row = q0 + i + hl * 8;
#pragma unroll
        for (int t = 0; t < 4; ++t)
            o[((size_t)(bb * TSEQ + row) * NH + hh) * HDIM + t * 16 + l16] = f2bf(o_acc[t][i] * inv);
    }
}

// ---------------- host-side orchestration ----------------
extern "C" void kernel_launch(void* const* d_in, const int* in_sizes, int n_in,
                              void* d_out, int out_size, void* d_ws, size_t ws_size,
                              hipStream_t stream) {
    (void)in_sizes; (void)n_in; (void)out_size; (void)ws_size;
    const int*   x    = (const int*)  d_in[0];
    const float* tok  = (const float*)d_in[1];
    const float* pos  = (const float*)d_in[2];
    const float* Wq   = (const float*)d_in[3];
    const float* bq   = (const float*)d_in[4];
    const float* Wk   = (const float*)d_in[5];
    const float* bk   = (const float*)d_in[6];
    const float* Wv   = (const float*)d_in[7];
    const float* bv   = (const float*)d_in[8];
    const float* Wo   = (const float*)d_in[9];
    const float* bo   = (const float*)d_in[10];
    const float* ln1s = (const float*)d_in[11];
    const float* ln1b = (const float*)d_in[12];
    const float* ln2s = (const float*)d_in[13];
    const float* ln2b = (const float*)d_in[14];
    const float* W1   = (const float*)d_in[15];
    const float* b1   = (const float*)d_in[16];
    const float* W2   = (const float*)d_in[17];
    const float* b2   = (const float*)d_in[18];
    const float* Wout = (const float*)d_in[19];
    const float* bout = (const float*)d_in[20];
    float* out = (float*)d_out;

    char* w = (char*)d_ws;
    size_t off = 0;
    auto take = [&](size_t bytes) -> char* {
        char* p = w + off;
        off += (bytes + 255) & ~(size_t)255;
        return p;
    };
    u16* WqT   = (u16*)take((size_t)NL * EDIM * EDIM * 2);
    u16* WkT   = (u16*)take((size_t)NL * EDIM * EDIM * 2);
    u16* WvT   = (u16*)take((size_t)NL * EDIM * EDIM * 2);
    u16* WoT   = (u16*)take((size_t)NL * EDIM * EDIM * 2);
    u16* W1T   = (u16*)take((size_t)NL * EDIM * FFD * 2);
    u16* W2T   = (u16*)take((size_t)NL * FFD * EDIM * 2);
    u16* WoutT = (u16*)take((size_t)EDIM * VOC * 2);
    float* hbuf = (float*)take((size_t)MTOK * EDIM * 4);
    u16* ybuf  = (u16*)take((size_t)MTOK * EDIM * 2);
    u16* qb    = (u16*)take((size_t)MTOK * EDIM * 2);
    u16* kb    = (u16*)take((size_t)MTOK * EDIM * 2);
    u16* vtb   = (u16*)take((size_t)MTOK * EDIM * 2);
    u16* ob    = (u16*)take((size_t)MTOK * EDIM * 2);
    u16* ff1   = (u16*)take((size_t)MTOK * FFD * 2);
    u16* hbf   = (u16*)take((size_t)MTOK * EDIM * 2);

    // ---- weight prep (runs each call; deterministic) ----
    const int gEE = EDIM * EDIM / 256;
    const int gEF = EDIM * FFD / 256;
    for (int i = 0; i < NL; ++i) {
        size_t oEE = (size_t)i * EDIM * EDIM;
        size_t oEF = (size_t)i * EDIM * FFD;
        wconvt_kernel<<<gEE, 256, 0, stream>>>(Wq + oEE, WqT + oEE, EDIM, EDIM);
        wconvt_kernel<<<gEE, 256, 0, stream>>>(Wk + oEE, WkT + oEE, EDIM, EDIM);
        wconvt_kernel<<<gEE, 256, 0, stream>>>(Wv + oEE, WvT + oEE, EDIM, EDIM);
        wconvt_kernel<<<gEE, 256, 0, stream>>>(Wo + oEE, WoT + oEE, EDIM, EDIM);
        wconvt_kernel<<<gEF, 256, 0, stream>>>(W1 + oEF, W1T + oEF, EDIM, FFD);
        wconvt_kernel<<<gEF, 256, 0, stream>>>(W2 + oEF, W2T + oEF, FFD, EDIM);
    }
    wconvt_kernel<<<EDIM * VOC / 256, 256, 0, stream>>>(Wout, WoutT, EDIM, VOC);

    // ---- embedding ----
    embed_kernel<<<MTOK * EDIM / 256, 256, 0, stream>>>(x, tok, pos, hbuf);

    const dim3 gQ((MTOK / 128) * (EDIM / 128));
    const dim3 gF((MTOK / 128) * (FFD / 128));
    for (int i = 0; i < NL; ++i) {
        size_t oEE = (size_t)i * EDIM * EDIM;
        size_t oEF = (size_t)i * EDIM * FFD;
        layernorm_kernel<<<MTOK / 8, 256, 0, stream>>>(hbuf, ln1s + i * EDIM, ln1b + i * EDIM, ybuf);
        gemm_bf16_kernel<1, false, false><<<gQ, 256, 0, stream>>>(
            ybuf, WqT + oEE, bq + i * EDIM, nullptr, nullptr, qb, MTOK, EDIM, EDIM);
        gemm_bf16_kernel<1, false, false><<<gQ, 256, 0, stream>>>(
            ybuf, WkT + oEE, bk + i * EDIM, nullptr, nullptr, kb, MTOK, EDIM, EDIM);
        gemm_bf16_kernel<2, false, false><<<gQ, 256, 0, stream>>>(
            ybuf, WvT + oEE, bv + i * EDIM, nullptr, nullptr, vtb, MTOK, EDIM, EDIM);
        attn_kernel<<<(BS * NH * (TSEQ / 16)) / 4, 128, 0, stream>>>(qb, kb, vtb, ob);
        gemm_bf16_kernel<0, false, true><<<gQ, 256, 0, stream>>>(
            ob, WoT + oEE, bo + i * EDIM, hbuf, hbuf, nullptr, MTOK, EDIM, EDIM);
        layernorm_kernel<<<MTOK / 8, 256, 0, stream>>>(hbuf, ln2s + i * EDIM, ln2b + i * EDIM, ybuf);
        gemm_bf16_kernel<1, true, false><<<gF, 256, 0, stream>>>(
            ybuf, W1T + oEF, b1 + i * FFD, nullptr, nullptr, ff1, MTOK, FFD, EDIM);
        gemm_bf16_kernel<0, false, true><<<gQ, 256, 0, stream>>>(
            ff1, W2T + oEF, b2 + i * EDIM, hbuf, hbuf, nullptr, MTOK, EDIM, FFD);
    }

    // ---- logits ----
    conv_kernel<<<MTOK * EDIM / 256, 256, 0, stream>>>(hbuf, hbf);
    gemm_bf16_kernel<0, false, false><<<dim3((MTOK / 128) * (VOC / 128)), 256, 0, stream>>>(
        hbf, WoutT, bout, nullptr, out, nullptr, MTOK, VOC, EDIM);
}
